// MultiHeadAttention_15092515078769
// MI455X (gfx1250) — matrix operand order repaired and checked by reading, not compiled, not run on hardware
//
#include <hip/hip_runtime.h>
#include <hip/hip_bf16.h>

// ---------------- problem constants ----------------
constexpr int B_ = 2, S_ = 2048, D_ = 1024, H_ = 16, DH_ = 64;

typedef __bf16 bf16;
typedef __attribute__((ext_vector_type(16))) __bf16 v16bf;
typedef __attribute__((ext_vector_type(8)))  __bf16 v8bf;
typedef __attribute__((ext_vector_type(8)))  float  v8f;

// ---------------- WMMA helpers ----------------
__device__ __forceinline__ v8f wmma_bf16(v16bf a, v16bf b, v8f c) {
  // D = A(16x32 bf16) * B(32x16 bf16) + C(16x16 f32)
  return __builtin_amdgcn_wmma_f32_16x16x32_bf16(
      /*neg_a=*/false, a, /*neg_b=*/false, b,
      /*c_mod=*/(short)0, c, /*reuse_a=*/false, /*reuse_b=*/false);
}

// B fragment: lane holds column N=lane%16, 16 consecutive K (half by lane>=16).
__device__ __forceinline__ v16bf ld16bf(const bf16* p) {
  v8bf lo = *reinterpret_cast<const v8bf*>(p);
  v8bf hi = *reinterpret_cast<const v8bf*>(p + 8);
  v16bf r;
#pragma unroll
  for (int i = 0; i < 8; ++i) { r[i] = lo[i]; r[i + 8] = hi[i]; }
  return r;
}

// A fragment: lane row M=lane%16; 16 consecutive K {2*koff..2*koff+15},
// koff = (lane>=16 ? 8 : 0). 'row' points at the 32-element K window of the row.
__device__ __forceinline__ v16bf ldfragA(const bf16* row, int koff) {
  v8bf lo = *reinterpret_cast<const v8bf*>(row + 2 * koff);
  v8bf hi = *reinterpret_cast<const v8bf*>(row + 2 * koff + 8);
  v16bf r;
#pragma unroll
  for (int i = 0; i < 8; ++i) { r[i] = lo[i]; r[i + 8] = hi[i]; }
  return r;
}

// Wait for all of this wave's outstanding async (global->LDS) transfers.
__device__ __forceinline__ void wait_async0() {
  asm volatile("s_wait_asynccnt 0x0" ::: "memory");
}

// ---------------- generic tiled WMMA GEMM with bias ----------------
// C[M,N] = A[M,K] @ Bw[K,N] + bias[N].  Block 128x128, 256 threads (8 waves 4x2),
// wave tile 32x64, BK=32. M,N multiples of 128; K multiple of 32.
template <typename AT, typename CT>
__global__ __launch_bounds__(256) void gemm_bias(const AT* __restrict__ A,
                                                 const float* __restrict__ Bw,
                                                 const float* __restrict__ bias,
                                                 CT* __restrict__ C,
                                                 int M, int N, int K) {
  __shared__ bf16 As[128 * 32];   // [m][k]
  __shared__ bf16 Bs[128 * 32];   // [n][k]  (transposed tile: K contiguous per column)

  const int tid  = threadIdx.x;
  const int lane = tid & 31;
  const int wid  = tid >> 5;
  const int wm   = wid >> 1;
  const int wn   = wid & 1;
  const int ln   = lane & 15;
  const int hi   = lane >> 4;
  const int koffA = hi * 8;
  const int koffB = hi * 16;
  const int bm = blockIdx.y * 128;
  const int bn = blockIdx.x * 128;

  v8f acc[2][4] = {};

  for (int k0 = 0; k0 < K; k0 += 32) {
    __syncthreads();
    { // stage A tile 128x32 (convert to bf16)
      int r = tid >> 1, kc = (tid & 1) * 16;
      const AT* src = A + (size_t)(bm + r) * K + k0 + kc;
      __builtin_prefetch(src + 32, 0, 0);
      bf16* dst = &As[r * 32 + kc];
#pragma unroll
      for (int i = 0; i < 16; ++i) dst[i] = (bf16)src[i];
    }
    { // stage B tile 32x128 transposed into Bs[n][k]
      int kk = tid >> 3, nc = (tid & 7) * 16;
      const float* src = Bw + (size_t)(k0 + kk) * N + bn + nc;
      __builtin_prefetch(src + (size_t)32 * N, 0, 0);
#pragma unroll
      for (int i = 0; i < 16; ++i) Bs[(nc + i) * 32 + kk] = (bf16)src[i];
    }
    __syncthreads();

    v16bf af[2], bfb[4];
#pragma unroll
    for (int mt = 0; mt < 2; ++mt)
      af[mt] = ldfragA(&As[(wm * 32 + mt * 16 + ln) * 32], koffA);
#pragma unroll
    for (int nt = 0; nt < 4; ++nt)
      bfb[nt] = ld16bf(&Bs[(wn * 64 + nt * 16 + ln) * 32 + koffB]);
#pragma unroll
    for (int mt = 0; mt < 2; ++mt)
#pragma unroll
      for (int nt = 0; nt < 4; ++nt)
        acc[mt][nt] = wmma_bf16(af[mt], bfb[nt], acc[mt][nt]);
  }

  // epilogue: C layout — lane ln = col, VGPR j -> row j + 8*hi
#pragma unroll
  for (int mt = 0; mt < 2; ++mt)
#pragma unroll
    for (int nt = 0; nt < 4; ++nt) {
      const int gc = bn + wn * 64 + nt * 16 + ln;
      const float bv = bias[gc];
#pragma unroll
      for (int j = 0; j < 8; ++j) {
        const int gr = bm + wm * 32 + mt * 16 + j + hi * 8;
        C[(size_t)gr * N + gc] = (CT)(acc[mt][nt][j] + bv);
      }
    }
}

// ---------------- V transpose: flat [B*S,D] -> [B,H,DH,S] ----------------
__global__ __launch_bounds__(256) void transpose_v(const bf16* __restrict__ Vb,
                                                   bf16* __restrict__ Vt) {
  size_t i = (size_t)blockIdx.x * 256 + threadIdx.x; // over B*H*DH*S, coalesced writes
  int s = (int)(i & (size_t)(S_ - 1));
  size_t t = i >> 11;               // /S_
  int d = (int)(t & (DH_ - 1));
  t >>= 6;                          // /DH_
  int h = (int)(t & (H_ - 1));
  int b = (int)(t >> 4);            // /H_
  Vt[i] = Vb[((size_t)(b * S_ + s)) * D_ + h * DH_ + d];
}

// ---------------- fused attention ----------------
// One workgroup = 32 query rows of one (b,h). Two-pass exact softmax; attn touches
// HBM exactly once (537 MB floor, stored non-temporal so K/V stay L2-resident).
// K tiles double-buffered via async global->LDS DMA (ASYNCcnt).

// Async-stage one 128x64 bf16 K block: 256 threads x 64B, ASYNCcnt-tracked.
// Instruction offset is added to BOTH the LDS and global addresses (ISA 15.18.3).
__device__ __forceinline__ void stage_k_async(const bf16* __restrict__ Kp, bf16* Ks,
                                              int kb, int tid) {
  const int n = tid >> 1, c = (tid & 1) * 32;            // 32 elems = 64 B per thread
  const unsigned long long g =
      (unsigned long long)(Kp + (size_t)(kb * 128 + n) * D_ + c);
  const unsigned lds = (unsigned)(unsigned long long)(Ks + n * 64 + c);
  asm volatile(
      "global_load_async_to_lds_b128 %0, %1, off\n\t"
      "global_load_async_to_lds_b128 %0, %1, off offset:16\n\t"
      "global_load_async_to_lds_b128 %0, %1, off offset:32\n\t"
      "global_load_async_to_lds_b128 %0, %1, off offset:48"
      :: "v"(lds), "v"(g) : "memory");
}

__device__ __forceinline__ void scores_block(const bf16* Qs, const bf16* Ks, float* Pk,
                                             const float* __restrict__ mrow,
                                             int wid, int ln, int hi) {
  const float mv = mrow[wid * 16 + ln] * -1e9f;  // additive mask for this lane's column
#pragma unroll
  for (int mt = 0; mt < 2; ++mt) {
    v8f acc = {};
#pragma unroll
    for (int ks = 0; ks < 2; ++ks) {
      v16bf a = ldfragA(Qs + (mt * 16 + ln) * 64 + ks * 32, hi * 8);
      v16bf b = ld16bf(Ks + (wid * 16 + ln) * 64 + ks * 32 + hi * 16);
      acc = wmma_bf16(a, b, acc);
    }
#pragma unroll
    for (int j = 0; j < 8; ++j)
      Pk[(mt * 16 + j + hi * 8) * 128 + wid * 16 + ln] = acc[j] * 0.125f + mv;
  }
}

__global__ __launch_bounds__(256) void attn_fused(const bf16* __restrict__ Qg,
                                                  const bf16* __restrict__ Kg,
                                                  const bf16* __restrict__ Vt,
                                                  const float* __restrict__ mask,
                                                  float* __restrict__ attnOut,
                                                  bf16* __restrict__ ctxOut) {
  __shared__ bf16  Qs[32 * 64];        // 4 KB
  __shared__ bf16  Ksb[2][128 * 64];   // 2 x 16 KB, async double buffer
  __shared__ float Pk[32 * 128];       // 16 KB raw score tile
  __shared__ bf16  Pb[32 * 128];       // 8 KB softmaxed bf16 tile for ctx WMMA
  __shared__ float red[256];           // 1 KB reduction scratch
  __shared__ float bM[32];             // per-row broadcast max
  __shared__ float rowM[32], rowInv[32];

  const int tid  = threadIdx.x;
  const int lane = tid & 31;
  const int wid  = tid >> 5;
  const int ln   = lane & 15;
  const int hi   = lane >> 4;
  const int strip = blockIdx.x;           // 0..S/32-1
  const int bh    = blockIdx.y;           // 0..B*H-1
  const int b = bh >> 4, h = bh & 15;
  const int s0 = strip * 32;

  const bf16* Qp = Qg + ((size_t)(b * S_ + s0)) * D_ + h * DH_;
  const bf16* Kp = Kg + ((size_t)b * S_) * D_ + h * DH_;
  const bf16* Vp = Vt + ((size_t)bh * DH_) * S_;   // [d][s], s contiguous
  const float* mp = mask + (size_t)b * S_;
  float* ap = attnOut + ((size_t)bh * S_ + s0) * S_;

  { // load Q strip 32x64
    int r = tid >> 3, c = (tid & 7) * 8;
    *reinterpret_cast<v8bf*>(&Qs[r * 64 + c]) =
        *reinterpret_cast<const v8bf*>(Qp + (size_t)r * D_ + c);
  }
  const int rr = tid >> 3, gg = tid & 7;   // 8 threads per row for reductions

  // ---------- pass 1: online row stats (max / sum-of-exp) ----------
  float m_r = -3.0e38f, l_r = 0.f;          // valid for tid<32 (row = tid)
  stage_k_async(Kp, Ksb[0], 0, tid);        // prefetch kb=0
  for (int kb = 0; kb < 16; ++kb) {
    bf16* cur = Ksb[kb & 1];
    bf16* nxt = Ksb[(kb & 1) ^ 1];
    wait_async0();
    __syncthreads();
    if (kb + 1 < 16) stage_k_async(Kp, nxt, kb + 1, tid);  // overlap DMA with compute
    scores_block(Qs, cur, Pk, mp + kb * 128, wid, ln, hi);
    __syncthreads();
    { // partial max over 16 cols per thread
      const float* pr = &Pk[rr * 128 + gg * 16];
      float mloc = pr[0];
#pragma unroll
      for (int i = 1; i < 16; ++i) mloc = fmaxf(mloc, pr[i]);
      red[tid] = mloc;
    }
    __syncthreads();
    if (tid < 32) {
      float mnew = m_r;
#pragma unroll
      for (int j = 0; j < 8; ++j) mnew = fmaxf(mnew, red[tid * 8 + j]);
      bM[tid] = mnew;
    }
    __syncthreads();
    { // partial sum of exp against the new row max
      const float mnew = bM[rr];
      const float* pr = &Pk[rr * 128 + gg * 16];
      float s = 0.f;
#pragma unroll
      for (int i = 0; i < 16; ++i) s += __expf(pr[i] - mnew);
      red[tid] = s;
    }
    __syncthreads();
    if (tid < 32) { // online combine (row leader)
      const float mnew = bM[tid];
      float s = l_r * __expf(m_r - mnew);
#pragma unroll
      for (int j = 0; j < 8; ++j) s += red[tid * 8 + j];
      m_r = mnew; l_r = s;
    }
  }
  __syncthreads();
  if (tid < 32) { rowM[tid] = m_r; rowInv[tid] = 1.0f / l_r; }

  // ---------- pass 2: attn write (once, non-temporal) + ctx WMMA ----------
  const int mtW = wid >> 2, ntW = wid & 3;   // ctx tile per wave: 2x4 grid of 16x16
  v8f cacc = {};
  stage_k_async(Kp, Ksb[0], 0, tid);
  for (int kb = 0; kb < 16; ++kb) {
    bf16* cur = Ksb[kb & 1];
    bf16* nxt = Ksb[(kb & 1) ^ 1];
    wait_async0();
    __syncthreads();
    if (kb + 1 < 16) stage_k_async(Kp, nxt, kb + 1, tid);
    scores_block(Qs, cur, Pk, mp + kb * 128, wid, ln, hi);
    __syncthreads();
    { // softmax-apply: Pk -> attn (HBM, NT store: write-once, never re-read) + Pb
#pragma unroll
      for (int i = 0; i < 16; ++i) {
        int idx = tid + i * 256;
        int r = idx >> 7, c = idx & 127;
        float p = __expf(Pk[idx] - rowM[r]) * rowInv[r];
        __builtin_nontemporal_store(p, &ap[(size_t)r * S_ + kb * 128 + c]);
        Pb[idx] = (bf16)p;
      }
    }
    __syncthreads();
    // ctx: C[32,64] += P[32,128] @ V[128,64]; B-fragments contiguous from Vt (L2-hot)
#pragma unroll
    for (int ks = 0; ks < 4; ++ks) {
      v16bf a = ldfragA(&Pb[(mtW * 16 + ln) * 128 + ks * 32], hi * 8);
      const bf16* vptr = Vp + (size_t)(ntW * 16 + ln) * S_ + kb * 128 + ks * 32 + hi * 16;
      v16bf bv = ld16bf(vptr);
      cacc = wmma_bf16(a, bv, cacc);
    }
  }

  { // write ctx tile (merge heads): [b, s0+m, h*DH + d] as bf16
    const int d = ntW * 16 + ln;
#pragma unroll
    for (int j = 0; j < 8; ++j) {
      const int m = mtW * 16 + j + hi * 8;
      ctxOut[((size_t)(b * S_ + s0 + m)) * D_ + h * DH_ + d] = (bf16)cacc[j];
    }
  }
}

// ---------------- host ----------------
extern "C" void kernel_launch(void* const* d_in, const int* in_sizes, int n_in,
                              void* d_out, int out_size, void* d_ws, size_t ws_size,
                              hipStream_t stream) {
  const float* q    = (const float*)d_in[0];
  const float* k    = (const float*)d_in[1];
  const float* v    = (const float*)d_in[2];
  const float* mask = (const float*)d_in[3];
  const float* wq_w = (const float*)d_in[4];
  const float* wq_b = (const float*)d_in[5];
  const float* wk_w = (const float*)d_in[6];
  const float* wk_b = (const float*)d_in[7];
  const float* wv_w = (const float*)d_in[8];
  const float* wv_b = (const float*)d_in[9];
  const float* wo_w = (const float*)d_in[10];
  const float* wo_b = (const float*)d_in[11];

  float* out  = (float*)d_out;
  float* attn = out + (size_t)B_ * S_ * D_;   // tuple output: [out | attn]

  const size_t bsd = (size_t)B_ * S_ * D_;    // 4M elements
  bf16* Qb  = (bf16*)d_ws;                    // needs 5 * bsd * 2 = 40 MB workspace
  bf16* Kb  = Qb + bsd;
  bf16* Vb  = Kb + bsd;
  bf16* Vt  = Vb + bsd;
  bf16* Ctx = Vt + bsd;

  dim3 blk(256);
  dim3 gproj(D_ / 128, (B_ * S_) / 128);      // (8, 32)

  gemm_bias<float, bf16><<<gproj, blk, 0, stream>>>(q, wq_w, wq_b, Qb, B_ * S_, D_, D_);
  gemm_bias<float, bf16><<<gproj, blk, 0, stream>>>(k, wk_w, wk_b, Kb, B_ * S_, D_, D_);
  gemm_bias<float, bf16><<<gproj, blk, 0, stream>>>(v, wv_w, wv_b, Vb, B_ * S_, D_, D_);

  transpose_v<<<(unsigned)(bsd / 256), blk, 0, stream>>>(Vb, Vt);

  dim3 gattn(S_ / 32, B_ * H_);               // (64, 32)
  attn_fused<<<gattn, blk, 0, stream>>>(Qb, Kb, Vt, mask, attn, Ctx);

  gemm_bias<bf16, float><<<gproj, blk, 0, stream>>>(Ctx, wo_w, wo_b, out, B_ * S_, D_, D_);
}
